// HybridQuantumClassifier_65481071395377
// MI455X (gfx1250) — matrix operand order repaired and checked by reading, not compiled, not run on hardware
//
#include <hip/hip_runtime.h>
#include <math.h>

// ---------------------------------------------------------------------------
// HybridQuantumClassifier fused pipeline for MI455X (gfx1250, wave32, WMMA)
//
// K2: per block, the shared j-tile (xn rows jb..jb+31, 8KB; x^T cols jb..jb+31,
// 8KB) is staged into LDS with global_load_async_to_lds_b128 (ASYNCcnt),
// double-buffered so stage k+1's DMA overlaps stage k's 17 WMMAs.
// sched_group_barrier pins a DS-read/WMMA interleave so the waitcnt pass
// emits partial s_wait_dscnt (load c+2 in flight under WMMA c).
// ---------------------------------------------------------------------------

#define N_ROWS   16384
#define F_DIM    128
#define H1_DIM   64
#define H2_DIM   32
#define OUT_DIM  2
#define COS_TH   0.94868329805f   // sqrt(0.9): |cos| >= sqrt(FID_THRESH)
#define BN_EPS   1e-5f

#define SGB_WMMA 0x008            // MFMA/WMMA scheduling class
#define SGB_DSRD 0x100            // DS read scheduling class

typedef __attribute__((ext_vector_type(16))) __bf16 v16bf;
typedef __attribute__((ext_vector_type(8)))  __bf16 v8bf;
typedef __attribute__((ext_vector_type(8)))  float  v8f;

// LDS tile geometry (bf16 elements). Rows padded for bank-conflict-free reads:
//  lA : 32 rows x 136 elems (272B stride -> lane l starts at bank 4*l)
//  lB : 128 rows x 40 elems (80B stride)
#define LA_STRIDE  136
#define LB_STRIDE  40
#define LB_BASE    (32 * LA_STRIDE)                 // 4352 elems
#define S_ELEMS    (LB_BASE + 128 * LB_STRIDE)      // 9472 elems per stage

__device__ __forceinline__ v8f wmma_bf16(v16bf a, v16bf b, v8f c) {
  return __builtin_amdgcn_wmma_f32_16x16x32_bf16(
      false, a, false, b, (short)0, c, false, false);
}

__device__ __forceinline__ unsigned lds_addr(const void* p) {
  // generic -> LDS addrspace -> 32-bit byte offset (VGPR operand for async DMA)
  return (unsigned)(size_t)(__attribute__((address_space(3))) const char*)p;
}

__device__ __forceinline__ void async_copy_b128(unsigned lds_off, const void* g) {
  asm volatile("global_load_async_to_lds_b128 %0, %1, off"
               :: "v"(lds_off), "v"((unsigned long long)(size_t)g)
               : "memory");
}

// ---------------------------------------------------------------- K1: prep
__global__ __launch_bounds__(256)
void hqc_prep_kernel(const float* __restrict__ x,
                     __bf16* __restrict__ xn,
                     __bf16* __restrict__ xpT) {
  const int row = blockIdx.x * blockDim.x + threadIdx.x;
  if (row >= N_ROWS) return;
  float v[F_DIM];
  float ss = 0.f;
#pragma unroll 8
  for (int f = 0; f < F_DIM; ++f) {
    v[f] = x[(size_t)row * F_DIM + f];
    ss = fmaf(v[f], v[f], ss);
  }
  const float inv = 1.0f / (sqrtf(ss) + 1e-12f);
#pragma unroll 8
  for (int f = 0; f < F_DIM; ++f) {
    xn[(size_t)row * F_DIM + f] = (__bf16)(v[f] * inv);
    xpT[(size_t)f * N_ROWS + row] = (__bf16)v[f];    // coalesced across rows
  }
}

// -------------------------------------------------- K2: fused gram+aggregate
__device__ __forceinline__ void stage_tiles(__bf16* smbase,
                                            const __bf16* __restrict__ xn,
                                            const __bf16* __restrict__ xpT,
                                            int jb, int tid) {
  // lA: xn rows jb..jb+31 (32 x 256B) as 512 x 16B chunks
#pragma unroll
  for (int q = 0; q < 4; ++q) {
    const int g = q * 128 + tid;
    const int row = g >> 4, c = g & 15;
    async_copy_b128(lds_addr(smbase + row * LA_STRIDE) + c * 16,
                    (const char*)(xn + (size_t)(jb + row) * F_DIM) + c * 16);
  }
  // lB: xpT[f][jb..jb+31] (128 x 64B) as 512 x 16B chunks
#pragma unroll
  for (int q = 0; q < 4; ++q) {
    const int g = q * 128 + tid;
    const int f = g >> 2, c = g & 3;
    async_copy_b128(lds_addr(smbase + LB_BASE + f * LB_STRIDE) + c * 16,
                    (const char*)(xpT + (size_t)f * N_ROWS + jb) + c * 16);
  }
}

__device__ __forceinline__ v16bf merge2(const __bf16* p, int gap_elems) {
  v8bf a = *(const v8bf*)p;
  v8bf b = *(const v8bf*)(p + gap_elems);
  v16bf r;
#pragma unroll
  for (int u = 0; u < 8; ++u) { r[u] = a[u]; r[u + 8] = b[u]; }
  return r;
}

__global__ __launch_bounds__(128)
void hqc_fused_graph_kernel(const float* __restrict__ x,
                            const __bf16* __restrict__ xn,
                            const __bf16* __restrict__ xpT,
                            float* __restrict__ agg) {
  __shared__ __align__(16) __bf16 sm[2 * S_ELEMS];   // ~37 KB, double-buffered

  const int tid  = threadIdx.x;
  const int lane = tid & 31;
  const int wave = tid >> 5;
  const int hi   = lane >> 4;
  const int lo   = lane & 15;
  const int ib   = blockIdx.x * 64 + wave * 16;      // wave's 16-row tile
  const int ig   = ib + lo;

  // Hoisted gram-B (Xn_I^T): elem u <-> (K = 8*hi + u (u<8), 16 + 8*hi + u-8 (u>=8), N = lo)
  v16bf Bg[4];
#pragma unroll
  for (int kc = 0; kc < 4; ++kc)
    Bg[kc] = merge2(xn + (size_t)(ib + lo) * F_DIM + kc * 32 + 8 * hi, 16);

  // Constant B-tile for the degree column (ones feature): col 0 = 1, rest 0
  v16bf Bdeg;
#pragma unroll
  for (int u = 0; u < 16; ++u) Bdeg[u] = (lo == 0) ? (__bf16)1.0f : (__bf16)0.0f;

  v8f acc[9];
#pragma unroll
  for (int c = 0; c < 9; ++c) acc[c] = (v8f){};

  const int diag_win = ib & ~31;    // uniform: the single jb that can hit the diagonal

  stage_tiles(sm, xn, xpT, 0, tid);

  for (int jb = 0; jb < N_ROWS; jb += 32) {
    const int st = (jb >> 5) & 1;
    const __bf16* cur = sm + st * S_ELEMS;

    if (jb + 32 < N_ROWS) {
      stage_tiles(sm + (st ^ 1) * S_ELEMS, xn, xpT, jb + 32, tid);
      asm volatile("s_wait_asynccnt 0x8" ::: "memory");  // current stage done
    } else {
      asm volatile("s_wait_asynccnt 0x0" ::: "memory");
    }
    __syncthreads();

    // ---- S^T tiles via 8 WMMAs (K=128); A feeds pipelined 1 deep
    const __bf16* aRow0 = cur + (size_t)lo * LA_STRIDE + 8 * hi;
    const __bf16* aRow1 = cur + (size_t)(16 + lo) * LA_STRIDE + 8 * hi;
    v8f s0 = (v8f){}, s1 = (v8f){};
    v16bf A0 = merge2(aRow0, 16);
    v16bf A1 = merge2(aRow1, 16);
#pragma unroll
    for (int kc = 0; kc < 3; ++kc) {
      v16bf A0n = merge2(aRow0 + (kc + 1) * 32, 16);
      v16bf A1n = merge2(aRow1 + (kc + 1) * 32, 16);
      s0 = wmma_bf16(A0, Bg[kc], s0);
      s1 = wmma_bf16(A1, Bg[kc], s1);
      A0 = A0n; A1 = A1n;
    }
    s0 = wmma_bf16(A0, Bg[3], s0);
    s1 = wmma_bf16(A1, Bg[3], s1);

    // Pin schedule: DSx8, (WMMAx2, DSx4)x2, WMMAx4 -> partial s_wait_dscnt
    __builtin_amdgcn_sched_group_barrier(SGB_DSRD, 8, 0);
    __builtin_amdgcn_sched_group_barrier(SGB_WMMA, 2, 0);
    __builtin_amdgcn_sched_group_barrier(SGB_DSRD, 4, 0);
    __builtin_amdgcn_sched_group_barrier(SGB_WMMA, 2, 0);
    __builtin_amdgcn_sched_group_barrier(SGB_DSRD, 4, 0);
    __builtin_amdgcn_sched_group_barrier(SGB_WMMA, 4, 0);

    // ---- threshold (|cos| >= sqrt(0.9)); diagonal killed only in its window
    v16bf M;
    if (jb == diag_win) {
#pragma unroll
      for (int u = 0; u < 16; ++u) {
        const float sv = (u < 8) ? s0[u] : s1[u - 8];
        const int   jg = jb + u + 8 * hi + ((u & 8) ? 8 : 0);
        M[u] = (__bf16)(((__builtin_fabsf(sv) >= COS_TH) && (jg != ig)) ? 1.0f : 0.0f);
      }
    } else {
#pragma unroll
      for (int u = 0; u < 16; ++u) {
        const float sv = (u < 8) ? s0[u] : s1[u - 8];
        M[u] = (__bf16)((__builtin_fabsf(sv) >= COS_TH) ? 1.0f : 0.0f);
      }
    }

    // ---- neigh_sum: mask(16x32) @ X_J(32x128); B2 feeds pipelined 1 deep
    const __bf16* bBase = cur + LB_BASE + (size_t)lo * LB_STRIDE + 8 * hi;
    v16bf B2 = merge2(bBase, 16);
#pragma unroll
    for (int c = 0; c < 7; ++c) {
      v16bf B2n = merge2(bBase + (size_t)(c + 1) * 16 * LB_STRIDE, 16);
      acc[c] = wmma_bf16(M, B2, acc[c]);
      B2 = B2n;
    }
    acc[7] = wmma_bf16(M, B2, acc[7]);
    acc[8] = wmma_bf16(M, Bdeg, acc[8]);

    // Pin schedule: DSx4, (WMMAx1, DSx2)x6, WMMAx3 -> two B2 bufs in flight
    __builtin_amdgcn_sched_group_barrier(SGB_DSRD, 4, 0);
#pragma unroll
    for (int c = 0; c < 6; ++c) {
      __builtin_amdgcn_sched_group_barrier(SGB_WMMA, 1, 0);
      __builtin_amdgcn_sched_group_barrier(SGB_DSRD, 2, 0);
    }
    __builtin_amdgcn_sched_group_barrier(SGB_WMMA, 3, 0);

    __syncthreads();   // protect buffer before next stage overwrites it
  }

  // ---- epilogue: agg = deg>0 ? (x + neigh/max(deg,1))*0.5 : x
#pragma unroll
  for (int r = 0; r < 8; ++r) {
    const float deg  = __shfl(acc[8][r], lane & 16, 32);  // broadcast col 0
    const int   row  = ib + r + 8 * hi;
    const float invd = 1.0f / fmaxf(deg, 1.0f);
#pragma unroll
    for (int c = 0; c < 8; ++c) {
      const int f = c * 16 + lo;
      const float xv = x[(size_t)row * F_DIM + f];
      const float nm = acc[c][r] * invd;
      agg[(size_t)row * F_DIM + f] = (deg > 0.0f) ? (xv + nm) * 0.5f : xv;
    }
  }
}

// ---------------------------------------------------------------- K3: MLP
__global__ __launch_bounds__(128)
void hqc_mlp_kernel(const float* __restrict__ agg,
                    const float* __restrict__ W1, const float* __restrict__ b1,
                    const float* __restrict__ W2, const float* __restrict__ b2,
                    const float* __restrict__ W3, const float* __restrict__ b3,
                    float* __restrict__ logits) {
  const int row = blockIdx.x * blockDim.x + threadIdx.x;
  if (row >= N_ROWS) return;
  float a[F_DIM];
#pragma unroll 8
  for (int k = 0; k < F_DIM; ++k) a[k] = agg[(size_t)row * F_DIM + k];

  float h[H1_DIM];
  for (int j = 0; j < H1_DIM; ++j) {
    float s = b1[j];
#pragma unroll 8
    for (int k = 0; k < F_DIM; ++k) s = fmaf(a[k], W1[k * H1_DIM + j], s);
    h[j] = fmaxf(s, 0.0f);
  }
  float g[H2_DIM];
  for (int j = 0; j < H2_DIM; ++j) {
    float s = b2[j];
#pragma unroll 8
    for (int k = 0; k < H1_DIM; ++k) s = fmaf(h[k], W2[k * H2_DIM + j], s);
    g[j] = fmaxf(s, 0.0f);
  }
#pragma unroll
  for (int c = 0; c < OUT_DIM; ++c) {
    float s = b3[c];
#pragma unroll 8
    for (int k = 0; k < H2_DIM; ++k) s = fmaf(g[k], W3[k * OUT_DIM + c], s);
    logits[(size_t)row * OUT_DIM + c] = s;
  }
}

// ------------------------------------------------------------ K4: BN stats
__global__ __launch_bounds__(256)
void hqc_bn_stats_kernel(const float* __restrict__ logits,
                         float* __restrict__ stats) {
  __shared__ float sh0[256], sh1[256], sq0[256], sq1[256];
  const int t = threadIdx.x;
  float s0 = 0.f, s1 = 0.f, q0 = 0.f, q1 = 0.f;
  for (int i = t; i < N_ROWS; i += 256) {
    const float a = logits[(size_t)i * 2 + 0];
    const float b = logits[(size_t)i * 2 + 1];
    s0 += a; s1 += b; q0 = fmaf(a, a, q0); q1 = fmaf(b, b, q1);
  }
  sh0[t] = s0; sh1[t] = s1; sq0[t] = q0; sq1[t] = q1;
  __syncthreads();
  for (int w = 128; w > 0; w >>= 1) {
    if (t < w) {
      sh0[t] += sh0[t + w]; sh1[t] += sh1[t + w];
      sq0[t] += sq0[t + w]; sq1[t] += sq1[t + w];
    }
    __syncthreads();
  }
  if (t == 0) {
    const float inv = 1.0f / (float)N_ROWS;
    const float m0 = sh0[0] * inv, m1 = sh1[0] * inv;
    stats[0] = m0;
    stats[1] = m1;
    stats[2] = sq0[0] * inv - m0 * m0;   // biased variance, matches reference
    stats[3] = sq1[0] * inv - m1 * m1;
  }
}

// ------------------------------------------------------------ K5: BN apply
__global__ __launch_bounds__(256)
void hqc_bn_apply_kernel(const float* __restrict__ logits,
                         const float* __restrict__ stats,
                         const float* __restrict__ gamma,
                         const float* __restrict__ beta,
                         float* __restrict__ out) {
  const int i = blockIdx.x * blockDim.x + threadIdx.x;
  if (i >= N_ROWS * OUT_DIM) return;
  const int c = i & 1;
  const float mu = stats[c], var = stats[2 + c];
  out[i] = (logits[i] - mu) * rsqrtf(var + BN_EPS) * gamma[c] + beta[c];
}

// ---------------------------------------------------------------------------
extern "C" void kernel_launch(void* const* d_in, const int* in_sizes, int n_in,
                              void* d_out, int out_size, void* d_ws, size_t ws_size,
                              hipStream_t stream) {
  const float* x     = (const float*)d_in[0];
  const float* W1    = (const float*)d_in[1];
  const float* b1    = (const float*)d_in[2];
  const float* W2    = (const float*)d_in[3];
  const float* b2    = (const float*)d_in[4];
  const float* W3    = (const float*)d_in[5];
  const float* b3    = (const float*)d_in[6];
  const float* gamma = (const float*)d_in[7];
  const float* beta  = (const float*)d_in[8];
  float* out = (float*)d_out;

  // workspace layout (all 256B aligned)
  char* ws = (char*)d_ws;
  const size_t off_xn   = 0;
  const size_t off_xpT  = off_xn  + (size_t)N_ROWS * F_DIM * 2;    // 4.0 MB
  const size_t off_agg  = off_xpT + (size_t)F_DIM * N_ROWS * 2;    // +4.0 MB
  const size_t off_log  = off_agg + (size_t)N_ROWS * F_DIM * 4;    // +8.0 MB
  const size_t off_stat = off_log + (size_t)N_ROWS * OUT_DIM * 4;  // +128 KB
  __bf16* xn    = (__bf16*)(ws + off_xn);
  __bf16* xpT   = (__bf16*)(ws + off_xpT);
  float*  agg   = (float*)(ws + off_agg);
  float*  logi  = (float*)(ws + off_log);
  float*  stats = (float*)(ws + off_stat);

  hqc_prep_kernel<<<N_ROWS / 256, 256, 0, stream>>>(x, xn, xpT);
  hqc_fused_graph_kernel<<<N_ROWS / 64, 128, 0, stream>>>(x, xn, xpT, agg);
  hqc_mlp_kernel<<<N_ROWS / 128, 128, 0, stream>>>(agg, W1, b1, W2, b2, W3, b3, logi);
  hqc_bn_stats_kernel<<<1, 256, 0, stream>>>(logi, stats);
  hqc_bn_apply_kernel<<<(N_ROWS * OUT_DIM) / 256, 256, 0, stream>>>(logi, stats, gamma, beta, out);
}
